// LongformerSelfAttention_66099546686030
// MI455X (gfx1250) — compile-verified, matrix operand order checked by reading, not run on hardware
//
#include <hip/hip_runtime.h>
#include <hip/hip_bf16.h>

typedef _Float16 v16h __attribute__((ext_vector_type(16)));
typedef _Float16 v8h  __attribute__((ext_vector_type(8)));
typedef float    v8f  __attribute__((ext_vector_type(8)));

#define S_LEN 4096
#define E_DIM 768
#define H_NUM 12
#define HD_DIM 64
#define B_NUM 2
#define W_RAD 256
#define G_NUM 32

static constexpr size_t NX = (size_t)B_NUM * S_LEN * E_DIM;  // 6291456 (8192x768)
static constexpr size_t NW = (size_t)E_DIM * E_DIM;          // 589824

// ---------------- WMMA helpers ----------------

__device__ __forceinline__ v8f wmma_f16(v16h a, v16h b, v8f c) {
  // D = A(16x32 f16) * B(32x16 f16) + C(16x16 f32)
  return __builtin_amdgcn_wmma_f32_16x16x32_f16(
      /*neg_a=*/false, a, /*neg_b=*/false, b,
      /*c_mod=*/(short)0, c, /*reuse_a=*/false, /*reuse_b=*/false);
}

// Build a 16-element f16 fragment from two contiguous 8-half (16B) groups.
// Per ISA layout: elements 0..7 = K kb0..kb0+7, elements 8..15 = K 16+kb0..16+kb0+7.
__device__ __forceinline__ v16h make_frag2(const _Float16* p0, const _Float16* p1) {
  v8h a = *(const v8h*)p0;
  v8h b = *(const v8h*)p1;
  v16h r;
#pragma unroll
  for (int i = 0; i < 8; ++i) { r[i] = a[i]; r[i + 8] = b[i]; }
  return r;
}
__device__ __forceinline__ v16h make_frag(const _Float16* p) { return make_frag2(p, p + 16); }

// Row-group reductions: rows live on 16-lane halves (lanes 0-15 rows 0-7, lanes 16-31 rows 8-15).
__device__ __forceinline__ float red_max16(float v) {
#pragma unroll
  for (int i = 1; i < 16; i <<= 1) v = fmaxf(v, __shfl_xor(v, i, 32));
  return v;
}
__device__ __forceinline__ float red_sum16(float v) {
#pragma unroll
  for (int i = 1; i < 16; i <<= 1) v += __shfl_xor(v, i, 32);
  return v;
}

// ---------------- Phase 1: casts ----------------

__global__ void cast_x_kernel(const float* __restrict__ x, _Float16* __restrict__ xh) {
  size_t i = (size_t)blockIdx.x * 256 + threadIdx.x;
  if (i < NX) xh[i] = (_Float16)x[i];
}

// Store W^T in f16: wt[id][n][k] = w_id[k][n]
__global__ void cast_w_kernel(const float* __restrict__ w0, const float* __restrict__ w1,
                              const float* __restrict__ w2, const float* __restrict__ w3,
                              const float* __restrict__ w4, const float* __restrict__ w5,
                              _Float16* __restrict__ wt) {
  size_t i = (size_t)blockIdx.x * 256 + threadIdx.x;
  int id = blockIdx.y;
  if (i >= NW) return;
  const float* w;
  switch (id) {
    case 0: w = w0; break; case 1: w = w1; break; case 2: w = w2; break;
    case 3: w = w3; break; case 4: w = w4; break; default: w = w5; break;
  }
  int n = (int)(i / E_DIM), k = (int)(i % E_DIM);
  wt[(size_t)id * NW + i] = (_Float16)w[(size_t)k * E_DIM + n];
}

// ---------------- Phase 2: projections (WMMA GEMM, 32x32 register block) ----------------
// Y_id = (X @ W_id + b_id) * scale_id, f16 output, row-major (8192, 768).
// ids: 0=q(*0.125) 1=k 2=v 3=kg 4=vg 5=qg(*0.125, only first G rows per batch)
__global__ __launch_bounds__(32) void proj_kernel(
    const _Float16* __restrict__ xh, const _Float16* __restrict__ wt,
    const float* __restrict__ b0, const float* __restrict__ b1,
    const float* __restrict__ b2, const float* __restrict__ b3,
    const float* __restrict__ b4, const float* __restrict__ b5,
    _Float16* __restrict__ outs) {
  const int nt = blockIdx.x;   // 0..23  (32 cols each)
  const int mt = blockIdx.y;   // 0..255 (32 rows each)
  const int id = blockIdx.z;   // 0..5
  if (id == 5 && (mt & 127) != 0) return;  // qg needs only rows s < 32 of each batch
  const int lane = threadIdx.x;
  const int ln = lane & 15;
  const int hi8 = (lane >> 4) << 3;
  const int kb0 = hi8;

  const _Float16* arow0 = xh + (size_t)(mt * 32 + ln) * E_DIM + kb0;
  const _Float16* arow1 = arow0 + (size_t)16 * E_DIM;
  const _Float16* brow0 = wt + (size_t)id * NW + (size_t)(nt * 32 + ln) * E_DIM + kb0;
  const _Float16* brow1 = brow0 + (size_t)16 * E_DIM;

  v8f acc00 = {}, acc01 = {}, acc10 = {}, acc11 = {};
#pragma unroll 2
  for (int k = 0; k < E_DIM; k += 32) {
    v16h A0 = make_frag(arow0 + k);
    v16h A1 = make_frag(arow1 + k);
    v16h B0 = make_frag(brow0 + k);
    v16h B1 = make_frag(brow1 + k);
    acc00 = wmma_f16(A0, B0, acc00);
    acc01 = wmma_f16(A0, B1, acc01);
    acc10 = wmma_f16(A1, B0, acc10);
    acc11 = wmma_f16(A1, B1, acc11);
  }
  const float* bias;
  switch (id) {
    case 0: bias = b0; break; case 1: bias = b1; break; case 2: bias = b2; break;
    case 3: bias = b3; break; case 4: bias = b4; break; default: bias = b5; break;
  }
  const float bv0 = bias[nt * 32 + ln];
  const float bv1 = bias[nt * 32 + 16 + ln];
  const float scale = (id == 0 || id == 5) ? 0.125f : 1.0f;
  _Float16* op = outs + (size_t)id * NX + (size_t)(mt * 32) * E_DIM + nt * 32 + ln;
#pragma unroll
  for (int r = 0; r < 8; ++r) {
    const size_t row0 = (size_t)(r + hi8) * E_DIM;
    const size_t row1 = row0 + (size_t)16 * E_DIM;
    op[row0]      = (_Float16)((acc00[r] + bv0) * scale);
    op[row0 + 16] = (_Float16)((acc01[r] + bv1) * scale);
    op[row1]      = (_Float16)((acc10[r] + bv0) * scale);
    op[row1 + 16] = (_Float16)((acc11[r] + bv1) * scale);
  }
}

// ---------------- Phase 3: transpose v / vg to [b][h][d][s] ----------------
__global__ void transpose_v_kernel(const _Float16* __restrict__ vh,
                                   const _Float16* __restrict__ vgh,
                                   _Float16* __restrict__ vT,
                                   _Float16* __restrict__ vgT) {
  size_t i = (size_t)blockIdx.x * 256 + threadIdx.x;
  if (i >= NX) return;
  const _Float16* src = blockIdx.y ? vgh : vh;
  _Float16* dst = blockIdx.y ? vgT : vT;
  int s = (int)(i % S_LEN);
  size_t t = i / S_LEN;
  int d = (int)(t % HD_DIM); t /= HD_DIM;
  int h = (int)(t % H_NUM);
  int b = (int)(t / H_NUM);
  dst[i] = src[(size_t)(b * S_LEN + s) * E_DIM + h * HD_DIM + d];
}

// ---------------- Phase 4: banded + global-key attention (flash) ----------------
// One wave per (b, h, 16-query tile). Keys: 2 tiles of global keys (0..31, no mask),
// then 34 tiles covering [qbase-W, qbase-W+544) with window validity + mask bias.
__global__ __launch_bounds__(32) void band_attn_kernel(
    const _Float16* __restrict__ qh, const _Float16* __restrict__ kh,
    const _Float16* __restrict__ vT, const int* __restrict__ amask,
    float* __restrict__ out) {
  const int lane = threadIdx.x;
  const int qt = blockIdx.x, h = blockIdx.y, b = blockIdx.z;
  const int qbase = qt * 16;
  const int ln = lane & 15;
  const int hi8 = (lane >> 4) << 3;
  const int kb0 = hi8;

  __shared__ __align__(16) _Float16 pbuf[16 * 40];  // probs transpose tile, pitch 40 halves

  const _Float16* qrow = qh + (size_t)(b * S_LEN + qbase + ln) * E_DIM + h * HD_DIM + kb0;
  const v16h qA0 = make_frag(qrow);
  const v16h qA1 = make_frag(qrow + 32);

  float m_r[8], l_r[8];
  v8f acc[4] = {};
#pragma unroll
  for (int r = 0; r < 8; ++r) { m_r[r] = -1e30f; l_r[r] = 0.0f; }

  for (int it = 0; it < 18; ++it) {  // 18 pairs of 16-key tiles = 576 key slots
    v8f s[2];
    int kp0 = 0;
#pragma unroll
    for (int sub = 0; sub < 2; ++sub) {
      const int tile = it * 2 + sub;
      const bool is_global = (tile < 2);
      const int kp = is_global ? (tile * 16 + ln)
                               : (qbase - W_RAD + (tile - 2) * 16 + ln);
      if (sub == 0) kp0 = kp - ln;
      const int kpc = min(max(kp, 0), S_LEN - 1);
      const _Float16* krow = kh + (size_t)(b * S_LEN + kpc) * E_DIM + h * HD_DIM + kb0;
      v16h kB0 = make_frag(krow);
      v16h kB1 = make_frag(krow + 32);
      v8f z = {};
      v8f sc = wmma_f16(qA0, kB0, z);
      sc = wmma_f16(qA1, kB1, sc);
      if (!is_global) {
        const float fbias = (amask[b * S_LEN + kpc] != 0) ? -10000.0f : 0.0f;
#pragma unroll
        for (int r = 0; r < 8; ++r) {
          const int qi = qbase + r + hi8;
          const bool valid = (kp >= 0) & (kp < S_LEN) & (kp >= qi - W_RAD) & (kp <= qi + W_RAD);
          sc[r] = valid ? (sc[r] + fbias) : -1e30f;
        }
      }
      s[sub] = sc;
    }
    // flash softmax update over this 32-key group
    float f_r[8];
#pragma unroll
    for (int r = 0; r < 8; ++r) {
      float tmax = red_max16(fmaxf(s[0][r], s[1][r]));
      float mn = fmaxf(m_r[r], tmax);
      f_r[r] = __expf(m_r[r] - mn);
      m_r[r] = mn;
      float p0 = __expf(s[0][r] - mn);
      float p1 = __expf(s[1][r] - mn);
      s[0][r] = p0; s[1][r] = p1;
      l_r[r] = l_r[r] * f_r[r] + red_sum16(p0 + p1);
    }
#pragma unroll
    for (int t = 0; t < 4; ++t)
#pragma unroll
      for (int r = 0; r < 8; ++r) acc[t][r] *= f_r[r];
    // transpose probs through LDS (D-layout -> A-layout)
#pragma unroll
    for (int r = 0; r < 8; ++r) {
      const int row = r + hi8;
      pbuf[row * 40 + ln] = (_Float16)s[0][r];
      pbuf[row * 40 + 16 + ln] = (_Float16)s[1][r];
    }
    __syncthreads();
    v16h pA = make_frag(&pbuf[ln * 40 + kb0]);
    // PV: B from vT[b][h][d][keys], keys contiguous; clamp 8-key groups (never straddle 0/S)
    int g0 = kp0 + kb0;
    int g1 = g0 + 16;
    g0 = (g0 < 0 || g0 > S_LEN - 8) ? 0 : g0;
    g1 = (g1 < 0 || g1 > S_LEN - 8) ? 0 : g1;
#pragma unroll
    for (int t = 0; t < 4; ++t) {
      const _Float16* vrow =
          vT + (size_t)((b * H_NUM + h) * HD_DIM + t * 16 + ln) * S_LEN;
      v16h vB = make_frag2(vrow + g0, vrow + g1);
      acc[t] = wmma_f16(pA, vB, acc[t]);
    }
    __syncthreads();
  }
  // epilogue
#pragma unroll
  for (int r = 0; r < 8; ++r) {
    const int qi = qbase + r + hi8;
    const float inv = 1.0f / l_r[r];
    float* op = out + (size_t)(b * S_LEN + qi) * E_DIM + h * HD_DIM + ln;
#pragma unroll
    for (int t = 0; t < 4; ++t) op[t * 16] = acc[t][r] * inv;
  }
}

// ---------------- Phase 5: global-token attention (rows < G), full-S flash ----------------
__global__ __launch_bounds__(32) void global_attn_kernel(
    const _Float16* __restrict__ qgh, const _Float16* __restrict__ kgh,
    const _Float16* __restrict__ vgT, float* __restrict__ out) {
  const int lane = threadIdx.x;
  const int gt = blockIdx.x, h = blockIdx.y, b = blockIdx.z;
  const int gbase = gt * 16;
  const int ln = lane & 15;
  const int hi8 = (lane >> 4) << 3;
  const int kb0 = hi8;

  __shared__ __align__(16) _Float16 pbuf[16 * 40];

  const _Float16* qrow = qgh + (size_t)(b * S_LEN + gbase + ln) * E_DIM + h * HD_DIM + kb0;
  const v16h qA0 = make_frag(qrow);
  const v16h qA1 = make_frag(qrow + 32);

  float m_r[8], l_r[8];
  v8f acc[4] = {};
#pragma unroll
  for (int r = 0; r < 8; ++r) { m_r[r] = -1e30f; l_r[r] = 0.0f; }

  for (int it = 0; it < S_LEN / 32; ++it) {
    const int kp0 = it * 32;
    v8f s[2];
#pragma unroll
    for (int sub = 0; sub < 2; ++sub) {
      const int kp = kp0 + sub * 16 + ln;
      const _Float16* krow = kgh + (size_t)(b * S_LEN + kp) * E_DIM + h * HD_DIM + kb0;
      v16h kB0 = make_frag(krow);
      v16h kB1 = make_frag(krow + 32);
      v8f z = {};
      v8f sc = wmma_f16(qA0, kB0, z);
      s[sub] = wmma_f16(qA1, kB1, sc);
    }
    float f_r[8];
#pragma unroll
    for (int r = 0; r < 8; ++r) {
      float tmax = red_max16(fmaxf(s[0][r], s[1][r]));
      float mn = fmaxf(m_r[r], tmax);
      f_r[r] = __expf(m_r[r] - mn);
      m_r[r] = mn;
      float p0 = __expf(s[0][r] - mn);
      float p1 = __expf(s[1][r] - mn);
      s[0][r] = p0; s[1][r] = p1;
      l_r[r] = l_r[r] * f_r[r] + red_sum16(p0 + p1);
    }
#pragma unroll
    for (int t = 0; t < 4; ++t)
#pragma unroll
      for (int r = 0; r < 8; ++r) acc[t][r] *= f_r[r];
#pragma unroll
    for (int r = 0; r < 8; ++r) {
      const int row = r + hi8;
      pbuf[row * 40 + ln] = (_Float16)s[0][r];
      pbuf[row * 40 + 16 + ln] = (_Float16)s[1][r];
    }
    __syncthreads();
    v16h pA = make_frag(&pbuf[ln * 40 + kb0]);
#pragma unroll
    for (int t = 0; t < 4; ++t) {
      const _Float16* vrow =
          vgT + (size_t)((b * H_NUM + h) * HD_DIM + t * 16 + ln) * S_LEN + kp0;
      v16h vB = make_frag2(vrow + kb0, vrow + kb0 + 16);
      acc[t] = wmma_f16(pA, vB, acc[t]);
    }
    __syncthreads();
  }
#pragma unroll
  for (int r = 0; r < 8; ++r) {
    const int g = gbase + r + hi8;
    const float inv = 1.0f / l_r[r];
    float* op = out + (size_t)(b * S_LEN + g) * E_DIM + h * HD_DIM + ln;
#pragma unroll
    for (int t = 0; t < 4; ++t) op[t * 16] = acc[t][r] * inv;
  }
}

// ---------------- Host launch ----------------

extern "C" void kernel_launch(void* const* d_in, const int* in_sizes, int n_in,
                              void* d_out, int out_size, void* d_ws, size_t ws_size,
                              hipStream_t stream) {
  (void)in_sizes; (void)n_in; (void)out_size; (void)ws_size;
  const float* x     = (const float*)d_in[0];
  const int*   amask = (const int*)d_in[1];
  // d_in[2] = layer_id (unused)
  const float* wq  = (const float*)d_in[3];  const float* bq  = (const float*)d_in[4];
  const float* wk  = (const float*)d_in[5];  const float* bk  = (const float*)d_in[6];
  const float* wv  = (const float*)d_in[7];  const float* bv  = (const float*)d_in[8];
  const float* wqg = (const float*)d_in[9];  const float* bqg = (const float*)d_in[10];
  const float* wkg = (const float*)d_in[11]; const float* bkg = (const float*)d_in[12];
  const float* wvg = (const float*)d_in[13]; const float* bvg = (const float*)d_in[14];
  float* out = (float*)d_out;

  // workspace layout (f16)
  _Float16* xh  = (_Float16*)d_ws;
  _Float16* wt  = xh + NX;          // 6 * NW (W^T, f16)
  _Float16* prj = wt + 6 * NW;      // 6 * NX: q,k,v,kg,vg,qg
  _Float16* qh  = prj + 0 * NX;
  _Float16* kh  = prj + 1 * NX;
  _Float16* vh  = prj + 2 * NX;
  _Float16* kgh = prj + 3 * NX;
  _Float16* vgh = prj + 4 * NX;
  _Float16* qgh = prj + 5 * NX;
  _Float16* vT  = prj + 6 * NX;     // [b][h][d][s]
  _Float16* vgT = vT + NX;

  cast_x_kernel<<<dim3((unsigned)((NX + 255) / 256)), 256, 0, stream>>>(x, xh);
  cast_w_kernel<<<dim3((unsigned)((NW + 255) / 256), 6), 256, 0, stream>>>(
      wq, wk, wv, wkg, wvg, wqg, wt);
  proj_kernel<<<dim3(E_DIM / 32, (B_NUM * S_LEN) / 32, 6), 32, 0, stream>>>(
      xh, wt, bq, bk, bv, bkg, bvg, bqg, prj);
  transpose_v_kernel<<<dim3((unsigned)((NX + 255) / 256), 2), 256, 0, stream>>>(
      vh, vgh, vT, vgT);
  band_attn_kernel<<<dim3(S_LEN / 16, H_NUM, B_NUM), 32, 0, stream>>>(
      qh, kh, vT, amask, out);
  global_attn_kernel<<<dim3(G_NUM / 16, H_NUM, B_NUM), 32, 0, stream>>>(
      qgh, kgh, vgT, out);
}
